// GAT_22247930594090
// MI455X (gfx1250) — matrix operand order repaired and checked by reading, not compiled, not run on hardware
//
#include <hip/hip_runtime.h>
#include <hip/hip_bf16.h>
#include <math.h>

// ---------------------------------------------------------------------------
// GAT 2-layer pipeline for MI455X (gfx1250, wave32, WMMA).
// GEMMs: LDS-staged f16 WMMA (f32 accumulate), branch-free main K-loop,
// clamp+select tail. Edge softmax/aggregate: wave32-per-edge scatter.
// ---------------------------------------------------------------------------

typedef __attribute__((ext_vector_type(16))) _Float16 v16h;
typedef __attribute__((ext_vector_type(8)))  _Float16 v8h;
typedef __attribute__((ext_vector_type(8)))  float    v8f;

#define NEG_SLOPE 0.2f

#define LDS_STRIDE 40   // halfs per row (80 bytes): 16B-aligned, bank-spread

// ---------------------------------------------------------------------------
// Tile fill: stage A[32 x 32k] and B[32k x 64] (as BsT[64][32k]) into LDS,
// converting f32 -> f16.  GUARD=true clamps addresses and zero-selects values
// for the K tail (no conditional loads -> no exec-mask branches).
// ---------------------------------------------------------------------------
template <bool GUARD>
__device__ __forceinline__ void gat_fill_tiles(
    const float* __restrict__ A, const float* __restrict__ B,
    _Float16* As, _Float16* BsT,
    int m0, int n0, int kc, int K, int Nc, int tid)
{
    // ---- As: 32 rows x 32 k. 128 threads, 8 halfs each (one b128 store).
    {
        const int r  = tid >> 2;          // 0..31
        const int k8 = (tid & 3) * 8;     // 0,8,16,24
        const float* ap = A + (size_t)(m0 + r) * K + kc + k8;
        _Float16 tmp[8];
        if (!GUARD) {
            __builtin_prefetch(ap + 32, 0, 3);
            const float4 v0 = *(const float4*)(ap);
            const float4 v1 = *(const float4*)(ap + 4);
            tmp[0] = (_Float16)v0.x; tmp[1] = (_Float16)v0.y;
            tmp[2] = (_Float16)v0.z; tmp[3] = (_Float16)v0.w;
            tmp[4] = (_Float16)v1.x; tmp[5] = (_Float16)v1.y;
            tmp[6] = (_Float16)v1.z; tmp[7] = (_Float16)v1.w;
        } else {
            const float* arow = A + (size_t)(m0 + r) * K;
            #pragma unroll
            for (int j = 0; j < 8; ++j) {
                const int k   = kc + k8 + j;
                const int kcl = (k < K) ? k : (K - 1);   // always-valid address
                const float v = arow[kcl];               // unconditional load
                tmp[j] = (_Float16)((k < K) ? v : 0.0f); // v_cndmask, no branch
            }
        }
        v8h pack;
        #pragma unroll
        for (int j = 0; j < 8; ++j) pack[j] = tmp[j];
        *(v8h*)(As + r * LDS_STRIDE + k8) = pack;        // ds_store_b128
    }
    // ---- BsT: 64 cols x 32 k (transposed). 128 threads, 16 halfs each.
    {
        const int k  = tid >> 2;          // 0..31
        const int cg = (tid & 3) * 16;    // 0,16,32,48
        float vals[16];
        if (!GUARD) {
            const float* bp = B + (size_t)(kc + k) * Nc + n0 + cg;
            const float4 v0 = *(const float4*)(bp);
            const float4 v1 = *(const float4*)(bp + 4);
            const float4 v2 = *(const float4*)(bp + 8);
            const float4 v3 = *(const float4*)(bp + 12);
            vals[0]=v0.x; vals[1]=v0.y; vals[2]=v0.z;  vals[3]=v0.w;
            vals[4]=v1.x; vals[5]=v1.y; vals[6]=v1.z;  vals[7]=v1.w;
            vals[8]=v2.x; vals[9]=v2.y; vals[10]=v2.z; vals[11]=v2.w;
            vals[12]=v3.x; vals[13]=v3.y; vals[14]=v3.z; vals[15]=v3.w;
        } else {
            const int kk  = kc + k;
            const int kcl = (kk < K) ? kk : (K - 1);
            const float* bp = B + (size_t)kcl * Nc + n0 + cg;
            #pragma unroll
            for (int j = 0; j < 16; ++j) {
                const float v = bp[j];                   // unconditional load
                vals[j] = (kk < K) ? v : 0.0f;
            }
        }
        #pragma unroll
        for (int j = 0; j < 16; ++j)
            BsT[(cg + j) * LDS_STRIDE + k] = (_Float16)vals[j];  // ds_store_b16
    }
}

// ---------------------------------------------------------------------------
// WMMA GEMM: C[M,Nc] = A[M,K] x B[K,Nc] + bias0[n] + bias1[n]
// Block: 128 threads = 4 waves. Block tile M=32, N=64; each wave owns one
// 16-column slice and computes two 16x16 tiles (rows 0-15, 16-31), reusing
// its B fragment. Requires M%32==0, Nc%64==0 (holds: 100000, 128/512).
// ---------------------------------------------------------------------------
__global__ __launch_bounds__(128) void gat_wmma_gemm(
    const float* __restrict__ A, const float* __restrict__ B,
    const float* __restrict__ bias0, const float* __restrict__ bias1,
    float* __restrict__ C, int M, int K, int Nc)
{
    __shared__ alignas(16) _Float16 As[32 * LDS_STRIDE];
    __shared__ alignas(16) _Float16 BsT[64 * LDS_STRIDE];

    const int tid  = threadIdx.x;
    const int lane = tid & 31;
    const int wave = tid >> 5;
    const bool hiH = lane >= 16;
    const int  sub = lane & 15;

    const int m0 = blockIdx.x * 32;
    const int n0 = blockIdx.y * 64;

    v8f acc0 = {};
    v8f acc1 = {};

    const int ao = hiH ? 8 : 0;    // A fragment half-offset within a row
    const int bo = hiH ? 8 : 0;    // B fragment K-offset
    const _Float16* arow0 = As + sub * LDS_STRIDE + ao;
    const _Float16* arow1 = As + (16 + sub) * LDS_STRIDE + ao;
    const _Float16* bcol  = BsT + (wave * 16 + sub) * LDS_STRIDE + bo;

    auto compute = [&]() {
        const v8h b_lo = *(const v8h*)(bcol);
        const v8h b_hi = *(const v8h*)(bcol + 16);
        const v16h b = __builtin_shufflevector(b_lo, b_hi,
            0,1,2,3,4,5,6,7,8,9,10,11,12,13,14,15);

        const v8h a0_lo = *(const v8h*)(arow0);
        const v8h a0_hi = *(const v8h*)(arow0 + 16);
        const v16h a0 = __builtin_shufflevector(a0_lo, a0_hi,
            0,1,2,3,4,5,6,7,8,9,10,11,12,13,14,15);

        const v8h a1_lo = *(const v8h*)(arow1);
        const v8h a1_hi = *(const v8h*)(arow1 + 16);
        const v16h a1 = __builtin_shufflevector(a1_lo, a1_hi,
            0,1,2,3,4,5,6,7,8,9,10,11,12,13,14,15);

        acc0 = __builtin_amdgcn_wmma_f32_16x16x32_f16(
            false, a0, false, b, (short)0, acc0, false, false);
        acc1 = __builtin_amdgcn_wmma_f32_16x16x32_f16(
            false, a1, false, b, (short)0, acc1, false, false);
    };

    int kc = 0;
    for (; kc + 32 <= K; kc += 32) {
        gat_fill_tiles<false>(A, B, As, BsT, m0, n0, kc, K, Nc, tid);
        __syncthreads();
        compute();
        __syncthreads();
    }
    if (kc < K) {   // K tail (K=300 case), runs once
        gat_fill_tiles<true>(A, B, As, BsT, m0, n0, kc, K, Nc, tid);
        __syncthreads();
        compute();
        __syncthreads();
    }

    // Epilogue. C/D layout: VGPR i = row (base+i) lanes 0-15, (base+8+i) lanes 16-31.
    const int col = n0 + wave * 16 + sub;
    const float bias = (bias0 ? bias0[col] : 0.0f) + (bias1 ? bias1[col] : 0.0f);
    const int rb0 = m0 + (hiH ? 8 : 0);
    const int rb1 = rb0 + 16;
    #pragma unroll
    for (int i = 0; i < 8; ++i)
        C[(size_t)(rb0 + i) * Nc + col] = acc0[i] + bias;
    #pragma unroll
    for (int i = 0; i < 8; ++i)
        C[(size_t)(rb1 + i) * Nc + col] = acc1[i] + bias;
}

// ---------------------------------------------------------------------------
// s_src[i] = h[i,:].a_src ; s_dst[i] = h[i,:].a_dst   (D % 4 == 0)
// ---------------------------------------------------------------------------
__global__ void gat_node_dots(const float* __restrict__ h,
                              const float* __restrict__ a_src,
                              const float* __restrict__ a_dst,
                              float* __restrict__ s_src, float* __restrict__ s_dst,
                              int N, int D)
{
    const int i = blockIdx.x * blockDim.x + threadIdx.x;
    if (i >= N) return;
    const float4* hr = (const float4*)(h + (size_t)i * D);
    const float4* as = (const float4*)a_src;
    const float4* ad = (const float4*)a_dst;
    float ss = 0.0f, sd = 0.0f;
    for (int d = 0; d < (D >> 2); ++d) {
        const float4 v = hr[d];
        const float4 s = as[d];
        const float4 t = ad[d];
        ss = fmaf(v.x, s.x, fmaf(v.y, s.y, fmaf(v.z, s.z, fmaf(v.w, s.w, ss))));
        sd = fmaf(v.x, t.x, fmaf(v.y, t.y, fmaf(v.z, t.z, fmaf(v.w, t.w, sd))));
    }
    s_src[i] = ss;
    s_dst[i] = sd;
}

__global__ void gat_seg_init(float* __restrict__ smax, float* __restrict__ denom, int N)
{
    const int i = blockIdx.x * blockDim.x + threadIdx.x;
    if (i >= N) return;
    smax[i]  = -INFINITY;
    denom[i] = 0.0f;
}

__device__ __forceinline__ void atomicMaxFloat(float* addr, float value)
{
    if (value >= 0.0f)
        atomicMax((int*)addr, __float_as_int(value));
    else
        atomicMin((unsigned int*)addr, __float_as_uint(value));
}

__global__ void gat_edge_score(const int* __restrict__ src, const int* __restrict__ dst,
                               const float* __restrict__ s_src, const float* __restrict__ s_dst,
                               float* __restrict__ score, float* __restrict__ smax, int E)
{
    const int e = blockIdx.x * blockDim.x + threadIdx.x;
    if (e >= E) return;
    float s = s_src[src[e]] + s_dst[dst[e]];
    s = (s > 0.0f) ? s : s * NEG_SLOPE;
    score[e] = s;
    atomicMaxFloat(&smax[dst[e]], s);
}

__global__ void gat_edge_exp(const int* __restrict__ dst,
                             float* __restrict__ score, const float* __restrict__ smax,
                             float* __restrict__ denom, int E)
{
    const int e = blockIdx.x * blockDim.x + threadIdx.x;
    if (e >= E) return;
    const int d = dst[e];
    const float v = __expf(score[e] - smax[d]);
    score[e] = v;
    atomicAdd(&denom[d], v);
}

// one wave32 per edge: out[dst,:] += alpha * h[src,:]
__global__ __launch_bounds__(256) void gat_edge_scatter(
    const int* __restrict__ src, const int* __restrict__ dst,
    const float* __restrict__ esc, const float* __restrict__ denom,
    const float* __restrict__ h, float* __restrict__ out, int E, int D)
{
    const int wavesPerBlock = blockDim.x >> 5;
    const int e = blockIdx.x * wavesPerBlock + (int)(threadIdx.x >> 5);
    if (e >= E) return;
    const int lane = threadIdx.x & 31;
    const int s = src[e];
    const int d = dst[e];
    const float alpha = esc[e] / (denom[d] + 1e-16f);
    const float* hs = h + (size_t)s * D;
    float* od = out + (size_t)d * D;
    for (int f = lane; f < D; f += 32)
        atomicAdd(&od[f], alpha * hs[f]);
}

__global__ void gat_relu(float* __restrict__ x, size_t n)
{
    const size_t i = (size_t)blockIdx.x * blockDim.x + threadIdx.x;
    if (i >= n) return;
    const float v = x[i];
    x[i] = v > 0.0f ? v : 0.0f;
}

// ---------------------------------------------------------------------------
extern "C" void kernel_launch(void* const* d_in, const int* in_sizes, int n_in,
                              void* d_out, int out_size, void* d_ws, size_t ws_size,
                              hipStream_t stream)
{
    const float* x       = (const float*)d_in[0];
    const int*   eidx    = (const int*)  d_in[1];
    const float* W1      = (const float*)d_in[2];
    const float* a_src1  = (const float*)d_in[3];
    const float* a_dst1  = (const float*)d_in[4];
    const float* b1      = (const float*)d_in[5];
    const float* lin1_W  = (const float*)d_in[6];
    const float* lin1_b  = (const float*)d_in[7];
    const float* W2      = (const float*)d_in[8];
    const float* a_src2  = (const float*)d_in[9];
    const float* a_dst2  = (const float*)d_in[10];
    const float* b2      = (const float*)d_in[11];
    const float* lin2_W  = (const float*)d_in[12];
    const float* lin2_b  = (const float*)d_in[13];
    float* out = (float*)d_out;

    const int D1   = in_sizes[3];            // 128
    const int D2   = in_sizes[9];            // 512
    const int D_IN = in_sizes[2] / D1;       // 300
    const int N    = in_sizes[0] / D_IN;     // 100000
    const int E    = in_sizes[1] / 2;        // 400000

    const int* src = eidx;                   // edge_index[0]
    const int* dst = eidx + E;               // edge_index[1]

    // Workspace layout (floats).
    float* ws = (float*)d_ws;
    size_t off = 0;
    float* h1    = ws + off; off += (size_t)N * D1;   // x@W1
    float* hbuf  = ws + off; off += (size_t)N * D1;   // lin1+b1, += agg, relu
    float* h2    = ws + off; off += (size_t)N * D2;   // h@W2
    float* s_src = ws + off; off += (size_t)N;
    float* s_dst = ws + off; off += (size_t)N;
    float* smax  = ws + off; off += (size_t)N;
    float* denom = ws + off; off += (size_t)N;
    float* esc   = ws + off; off += (size_t)E;
    (void)ws_size; (void)n_in; (void)out_size;

    const int TPB = 256;
    const int nodeBlocks = (N + TPB - 1) / TPB;
    const int edgeBlocks = (E + TPB - 1) / TPB;
    const int wavesPerBlock = TPB / 32;
    const int scatterBlocks = (E + wavesPerBlock - 1) / wavesPerBlock;

    // ---------------- Layer 1 ----------------
    {
        dim3 g1(N / 32, D1 / 64);
        gat_wmma_gemm<<<g1, 128, 0, stream>>>(x, W1, nullptr, nullptr, h1, N, D_IN, D1);
        gat_wmma_gemm<<<g1, 128, 0, stream>>>(x, lin1_W, lin1_b, b1, hbuf, N, D_IN, D1);

        gat_node_dots<<<nodeBlocks, TPB, 0, stream>>>(h1, a_src1, a_dst1, s_src, s_dst, N, D1);
        gat_seg_init<<<nodeBlocks, TPB, 0, stream>>>(smax, denom, N);
        gat_edge_score<<<edgeBlocks, TPB, 0, stream>>>(src, dst, s_src, s_dst, esc, smax, E);
        gat_edge_exp<<<edgeBlocks, TPB, 0, stream>>>(dst, esc, smax, denom, E);
        gat_edge_scatter<<<scatterBlocks, TPB, 0, stream>>>(src, dst, esc, denom, h1, hbuf, E, D1);

        const size_t n1 = (size_t)N * D1;
        gat_relu<<<(int)((n1 + TPB - 1) / TPB), TPB, 0, stream>>>(hbuf, n1);
    }

    // ---------------- Layer 2 ----------------
    {
        dim3 g2(N / 32, D2 / 64);
        gat_wmma_gemm<<<g2, 128, 0, stream>>>(hbuf, W2, nullptr, nullptr, h2, N, D1, D2);
        gat_wmma_gemm<<<g2, 128, 0, stream>>>(hbuf, lin2_W, lin2_b, b2, out, N, D1, D2);

        gat_node_dots<<<nodeBlocks, TPB, 0, stream>>>(h2, a_src2, a_dst2, s_src, s_dst, N, D2);
        gat_seg_init<<<nodeBlocks, TPB, 0, stream>>>(smax, denom, N);
        gat_edge_score<<<edgeBlocks, TPB, 0, stream>>>(src, dst, s_src, s_dst, esc, smax, E);
        gat_edge_exp<<<edgeBlocks, TPB, 0, stream>>>(dst, esc, smax, denom, E);
        gat_edge_scatter<<<scatterBlocks, TPB, 0, stream>>>(src, dst, esc, denom, h2, out, E, D2);
    }
}
